// Attention_3358664426137
// MI455X (gfx1250) — compile-verified, matrix operand order checked by reading, not run on hardware
//
#include <hip/hip_runtime.h>
#include <hip/hip_bf16.h>

typedef __attribute__((ext_vector_type(16))) _Float16 v16h;
typedef __attribute__((ext_vector_type(8)))  _Float16 v8h;
typedef __attribute__((ext_vector_type(8)))  float    v8f;
typedef __attribute__((ext_vector_type(4)))  unsigned int u32x4;
typedef __attribute__((ext_vector_type(8)))  int i32x8;
typedef __attribute__((ext_vector_type(4)))  int i32x4;

#define S_LEN   2048
#define P_LEN   2048
#define E_DIM   1024
#define H_NUM   16
#define D_DIM   64
#define KV_LEN  4096              // P + S
#define MASK_NEG (-10000.0f)

// Tensor Data Mover availability (device pass only; host pass parses fallback)
#if defined(__gfx1250__) && __has_builtin(__builtin_amdgcn_tensor_load_to_lds)
#define USE_TDM 1
#else
#define USE_TDM 0
#endif
#if __has_include(<hip/amd_detail/amd_gfx1250_TDM.h>)
#define TDM_6ARG 1                // therock-10.0 headers -> clang-23 6-arg builtin
#else
#define TDM_6ARG 0                // ROCm 7.2 clang-22 5-arg builtin
#endif

// ---------------------------------------------------------------------------
// WMMA helpers (CDNA5 wave32, 16x16x32 f16 -> f32)
// A-fragment (ISA 7.12.2): lane L<16 holds row L, K {k0..k0+7},{k0+16..k0+23};
// lanes 16..31 the +8 K offsets -> two contiguous 8-half chunks per lane from
// a row-major tile. B stored transposed loads with the identical pattern.
// ---------------------------------------------------------------------------
__device__ __forceinline__ v16h load_frag(const _Float16* p, int stride) {
  const int lane = threadIdx.x & 31;
  const int r    = lane & 15;
  const int hi   = lane >> 4;
  const _Float16* q = p + r * stride + hi * 8;
  v8h lo = *(const v8h*)(q);
  v8h hp = *(const v8h*)(q + 16);
  v16h out;
#pragma unroll
  for (int i = 0; i < 8; ++i) { out[i] = lo[i]; out[i + 8] = hp[i]; }
  return out;
}

__device__ __forceinline__ v8f wmma_f16(v16h a, v16h b, v8f c) {
  return __builtin_amdgcn_wmma_f32_16x16x32_f16(false, a, false, b, (short)0, c,
                                                false, false);
}

#if USE_TDM
// ---------------------------------------------------------------------------
// TDM 2D tile DMA: global (row-major, stride_elems halfs/row) -> LDS with
// hardware row padding. pad_interval code: DWORDs-per-row -> 0=2,1=4,2=8,
// 3=16,4=32...; pad_amount code: pad DWORDs minus 1. D# packing per ISA
// 8.3/8.4 (group0: count/lds/global/type=2; group1: sizes/strides/pads).
// Low 32 bits of a generic __shared__ pointer are the LDS byte offset
// (ISA 10.2 aperture mapping).
// ---------------------------------------------------------------------------
__device__ __forceinline__ void tdm_load_2d(const void* gptr,
                                            unsigned int lds_byte_off,
                                            int tile_d0_elems, int tile_d1_rows,
                                            int stride_elems,
                                            int pad_interval_code,
                                            int pad_amount_code) {
  unsigned long long ga = (unsigned long long)(uintptr_t)gptr;
  u32x4 g0;
  g0[0] = 1u;                                            // count=1, user mode
  g0[1] = lds_byte_off;                                  // lds_addr
  g0[2] = (unsigned int)ga;                              // global_addr[31:0]
  g0[3] = (unsigned int)((ga >> 32) & 0x01FFFFFFu) | (2u << 30);  // hi | type=2
  i32x8 g1;
  g1[0] = (1 << 16) | (1 << 20) |                        // data_size=2B, pad_en
          (pad_interval_code << 22) | (pad_amount_code << 25);
  const int tensor_d0 = stride_elems;                    // generous bounds (tile
  const int tensor_d1 = 1 << 20;                         //  always in range)
  g1[1] = (tensor_d0 & 0xFFFF) << 16;                    // tensor_dim0[15:0]
  g1[2] = ((tensor_d0 >> 16) & 0xFFFF) | ((tensor_d1 & 0xFFFF) << 16);
  g1[3] = ((tensor_d1 >> 16) & 0xFFFF) | (tile_d0_elems << 16);  // tile_dim0
  g1[4] = tile_d1_rows & 0xFFFF;                         // tile_dim1, dim2=0
  g1[5] = stride_elems;                                  // tensor_dim0_stride
  g1[6] = 0;
  g1[7] = 0;
  i32x4 z4 = {};
#if TDM_6ARG
  i32x8 z8 = {};
  __builtin_amdgcn_tensor_load_to_lds(g0, g1, z4, z4, z8, 0);
#else
  __builtin_amdgcn_tensor_load_to_lds(g0, g1, z4, z4, 0);
#endif
}
#endif // USE_TDM

// ---------------------------------------------------------------------------
// Kernel 1: precision conversion, weight transposes, past-KV staging
// ---------------------------------------------------------------------------
__global__ __launch_bounds__(256)
void prep_kernel(const float* __restrict__ x,
                 const float* __restrict__ w_attn,   // (E, 3E) row-major
                 const float* __restrict__ w_proj,   // (E, E) row-major
                 const float* __restrict__ past_k,   // (H, P, D)
                 const float* __restrict__ past_v,   // (H, P, D)
                 _Float16* __restrict__ xh,          // (S, E)
                 _Float16* __restrict__ wattnT,      // (3E, E) = w_attn^T
                 _Float16* __restrict__ wprojT,      // (E, E)  = w_proj^T
                 _Float16* __restrict__ k_h,         // (H, KV, D)
                 _Float16* __restrict__ v_h,         // (H, KV, D)
                 float* __restrict__ presK,          // (H, KV, D)
                 float* __restrict__ presV)          // (H, KV, D)
{
  const long N0 = (long)S_LEN * E_DIM;
  const long N1 = (long)3 * E_DIM * E_DIM;
  const long N2 = (long)E_DIM * E_DIM;
  const long N3 = (long)H_NUM * P_LEN * D_DIM;
  const long total = N0 + N1 + N2 + N3 + N3;
  const long stride = (long)gridDim.x * blockDim.x;
  for (long i = (long)blockIdx.x * blockDim.x + threadIdx.x; i < total; i += stride) {
    if (i < N0) {
      xh[i] = (_Float16)x[i];
    } else if (i < N0 + N1) {
      long idx = i - N0;                        // n * E + k
      long n = idx >> 10, k = idx & 1023;
      wattnT[idx] = (_Float16)w_attn[k * (3 * E_DIM) + n];
    } else if (i < N0 + N1 + N2) {
      long idx = i - N0 - N1;
      long n = idx >> 10, k = idx & 1023;
      wprojT[idx] = (_Float16)w_proj[k * E_DIM + n];
    } else if (i < N0 + N1 + N2 + N3) {
      long idx = i - N0 - N1 - N2;              // (h*P + p)*D + d
      long h = idx >> 17;
      long r = idx & ((1L << 17) - 1);
      long dst = (h << 18) + r;
      float v = past_k[idx];
      k_h[dst] = (_Float16)v;
      presK[dst] = v;
    } else {
      long idx = i - N0 - N1 - N2 - N3;
      long h = idx >> 17;
      long r = idx & ((1L << 17) - 1);
      long dst = (h << 18) + r;
      float v = past_v[idx];
      v_h[dst] = (_Float16)v;
      presV[dst] = v;
    }
  }
}

// ---------------------------------------------------------------------------
// Kernels 2 & 4: tiled WMMA GEMM  C = A(MxK) * Bt^T(KxN) + bias
//   mode 0: QKV epilogue (t = n0>>10 is block-uniform -> single scalar branch)
//   mode 1: plain f32 output
// 256 threads (8 waves), 128x128 tile, K-step 32.
// TDM path: double-buffered tensor_load_to_lds issued by wave 0 only (scalar
// branch via readfirstlane; TDM ignores EXEC so masking would not suppress it)
// overlapping the next K-slice DMA with this slice's WMMAs.
// ---------------------------------------------------------------------------
__global__ __launch_bounds__(256)
void gemm_kernel(const _Float16* __restrict__ A,
                 const _Float16* __restrict__ Bt,   // (N, K) row-major
                 const float* __restrict__ bias,
                 int K, int mode,
                 float* __restrict__ out_plain, int ldo,
                 _Float16* __restrict__ q_h,
                 _Float16* __restrict__ k_h,
                 _Float16* __restrict__ v_h,
                 float* __restrict__ presK,
                 float* __restrict__ presV)
{
  __shared__ _Float16 As[2][128 * 40];   // 32 K-halfs + 8 pad per row
  __shared__ _Float16 Bs[2][128 * 40];

  const int tid  = threadIdx.x;
  const int w    = tid >> 5;
  const int lane = tid & 31;
  const int m0 = blockIdx.x * 128;
  const int n0 = blockIdx.y * 128;
  const int wm = (w & 3) * 32;
  const int wn = (w >> 2) * 64;

  v8f acc[2][4] = {};

#if USE_TDM
  const bool issuer = (__builtin_amdgcn_readfirstlane(w) == 0);
  if (issuer) {
    // tile row = 32 halfs = 16 DWORDs (interval code 3); pad 4 DWORDs (code 3)
    tdm_load_2d(&A[(size_t)m0 * K], (unsigned int)(uintptr_t)&As[0][0],
                32, 128, K, 3, 3);
    tdm_load_2d(&Bt[(size_t)n0 * K], (unsigned int)(uintptr_t)&Bs[0][0],
                32, 128, K, 3, 3);
  }
#endif

  for (int k0 = 0; k0 < K; k0 += 32) {
    const int buf = (k0 >> 5) & 1;
#if USE_TDM
    __builtin_amdgcn_s_wait_tensorcnt(0);   // this slice's tiles landed
    __syncthreads();
    if (k0 + 32 < K && issuer) {            // overlap next DMA with compute
      tdm_load_2d(&A[(size_t)m0 * K + k0 + 32],
                  (unsigned int)(uintptr_t)&As[buf ^ 1][0], 32, 128, K, 3, 3);
      tdm_load_2d(&Bt[(size_t)n0 * K + k0 + 32],
                  (unsigned int)(uintptr_t)&Bs[buf ^ 1][0], 32, 128, K, 3, 3);
    }
#else
#pragma unroll
    for (int i = 0; i < 2; ++i) {
      int chunk = tid + i * 256;
      int row = chunk >> 2;
      int c8  = (chunk & 3) * 8;
      *(v8h*)&As[buf][row * 40 + c8] =
          *(const v8h*)&A[(size_t)(m0 + row) * K + k0 + c8];
      *(v8h*)&Bs[buf][row * 40 + c8] =
          *(const v8h*)&Bt[(size_t)(n0 + row) * K + k0 + c8];
    }
    if (k0 + 32 < K) {
      int row = tid >> 2, c8 = (tid & 3) * 8;
      __builtin_prefetch(&A[(size_t)(m0 + row) * K + k0 + 32 + c8], 0, 2);
      __builtin_prefetch(&Bt[(size_t)(n0 + row) * K + k0 + 32 + c8], 0, 2);
    }
    __syncthreads();
#endif

    v16h af[2], bf[4];
#pragma unroll
    for (int i = 0; i < 2; ++i) af[i] = load_frag(&As[buf][(wm + i * 16) * 40], 40);
#pragma unroll
    for (int j = 0; j < 4; ++j) bf[j] = load_frag(&Bs[buf][(wn + j * 16) * 40], 40);
#pragma unroll
    for (int i = 0; i < 2; ++i)
#pragma unroll
      for (int j = 0; j < 4; ++j) acc[i][j] = wmma_f16(af[i], bf[j], acc[i][j]);
    __syncthreads();
  }

  // Epilogue. C layout: element e -> row (e + 8*hi), col (lane & 15)
  const int r  = lane & 15;
  const int hi = lane >> 4;
  if (mode == 1) {
#pragma unroll
    for (int i = 0; i < 2; ++i)
#pragma unroll
      for (int j = 0; j < 4; ++j)
#pragma unroll
        for (int e = 0; e < 8; ++e) {
          int M = m0 + wm + i * 16 + e + 8 * hi;
          int N = n0 + wn + j * 16 + r;
          out_plain[(size_t)M * ldo + N] = acc[i][j][e] + bias[N];
        }
  } else {
    const int t = n0 >> 10;              // block-uniform: 0=q 1=k 2=v
    if (t == 0) {
#pragma unroll
      for (int i = 0; i < 2; ++i)
#pragma unroll
        for (int j = 0; j < 4; ++j)
#pragma unroll
          for (int e = 0; e < 8; ++e) {
            int M = m0 + wm + i * 16 + e + 8 * hi;
            int N = n0 + wn + j * 16 + r;
            int h = (N >> 6) & 15, d = N & 63;
            // fold softmax scale 1/sqrt(D)=0.125 into q
            q_h[((size_t)(h * S_LEN + M)) * D_DIM + d] =
                (_Float16)((acc[i][j][e] + bias[N]) * 0.125f);
          }
    } else {
      _Float16* dsth = (t == 1) ? k_h : v_h;
      float*    dstf = (t == 1) ? presK : presV;
#pragma unroll
      for (int i = 0; i < 2; ++i)
#pragma unroll
        for (int j = 0; j < 4; ++j)
#pragma unroll
          for (int e = 0; e < 8; ++e) {
            int M = m0 + wm + i * 16 + e + 8 * hi;
            int N = n0 + wn + j * 16 + r;
            int h = (N >> 6) & 15, d = N & 63;
            float val = acc[i][j][e] + bias[N];
            size_t dst = (size_t)h * KV_LEN * D_DIM + (size_t)(P_LEN + M) * D_DIM + d;
            dsth[dst] = (_Float16)val;
            dstf[dst] = val;
          }
    }
  }
}

// ---------------------------------------------------------------------------
// Kernel 3: flash-style causal attention.
// grid = (S/64, H), 128 threads = 4 waves; each wave owns 16 query rows.
// K tiles via double-buffered TDM (wave-0 issue, TENSORcnt + barrier);
// V tiles loaded transposed by the waves so P*V B-fragments are contiguous.
// ---------------------------------------------------------------------------
__global__ __launch_bounds__(128)
void attn_kernel(const _Float16* __restrict__ q_h,   // (H, S, D), pre-scaled
                 const _Float16* __restrict__ k_h,   // (H, KV, D)
                 const _Float16* __restrict__ v_h,   // (H, KV, D)
                 _Float16* __restrict__ a_h)         // (S, E)
{
  __shared__ _Float16 Ks[2][64 * 72];    // double-buffered keys (key x d)
  __shared__ _Float16 Vt[2][64 * 72];    // double-buffered V^T (d x key)
  __shared__ _Float16 Ps[4 * 16 * 72];   // per-wave probs (16 x 64)

  const int tid  = threadIdx.x;
  const int w    = tid >> 5;
  const int lane = tid & 31;
  const int r    = lane & 15;
  const int hi   = lane >> 4;
  const int h    = blockIdx.y;
  const int q0   = blockIdx.x * 64;
  const int qw   = q0 + w * 16;

  const _Float16* kh = k_h + (size_t)h * KV_LEN * D_DIM;
  const _Float16* vh = v_h + (size_t)h * KV_LEN * D_DIM;

  const _Float16* qbase = q_h + ((size_t)h * S_LEN + qw) * D_DIM;
  v16h qf0 = load_frag(qbase, D_DIM);
  v16h qf1 = load_frag(qbase + 32, D_DIM);

  float mrun[8], lrun[8];
#pragma unroll
  for (int e = 0; e < 8; ++e) { mrun[e] = -3.0e38f; lrun[e] = 0.0f; }
  v8f o[4] = {};

  const int nkb = (P_LEN + q0 + 64) >> 6;

#if USE_TDM
  const bool issuer = (__builtin_amdgcn_readfirstlane(w) == 0);
  if (issuer)
    // tile row = 64 halfs = 32 DWORDs (interval code 4); pad 4 DWORDs (code 3)
    tdm_load_2d(kh, (unsigned int)(uintptr_t)&Ks[0][0], 64, 64, D_DIM, 4, 3);
#endif

  for (int it = 0; it < nkb; ++it) {
    const int kb = it * 64;
    const int buf = it & 1;
    _Float16* KsB = &Ks[buf][0];
    _Float16* VtB = &Vt[buf][0];

    // V tile: cooperative transposed load (64x64 halfs, 4 chunks/thread)
#pragma unroll
    for (int c = 0; c < 4; ++c) {
      int chunk = tid + c * 128;
      int row = chunk >> 3;
      int c8  = (chunk & 7) * 8;
      v8h vv = *(const v8h*)&vh[(size_t)(kb + row) * D_DIM + c8];
#pragma unroll
      for (int x = 0; x < 8; ++x) VtB[(c8 + x) * 72 + row] = vv[x];
#if !USE_TDM
      v8h kv = *(const v8h*)&kh[(size_t)(kb + row) * D_DIM + c8];
      *(v8h*)&KsB[row * 72 + c8] = kv;
#endif
    }
#if USE_TDM
    __builtin_amdgcn_s_wait_tensorcnt(0);   // trivial for non-issuing waves
#endif
    __syncthreads();
#if USE_TDM
    if (it + 1 < nkb && issuer)             // overlap next DMA with compute
      tdm_load_2d(kh + (size_t)(kb + 64) * D_DIM,
                  (unsigned int)(uintptr_t)&Ks[buf ^ 1][0], 64, 64, D_DIM, 4, 3);
#endif

    // Scores: (16 x 64) = Q (16xD) * K_block^T
    v8f s[4] = {};
#pragma unroll
    for (int j = 0; j < 4; ++j) {
      v16h b0 = load_frag(&KsB[j * 16 * 72], 72);
      s[j] = wmma_f16(qf0, b0, s[j]);
      v16h b1 = load_frag(&KsB[j * 16 * 72 + 32], 72);
      s[j] = wmma_f16(qf1, b1, s[j]);
    }

    // Causal mask: key kb+j*16+r allowed iff <= q + P
#pragma unroll
    for (int j = 0; j < 4; ++j)
#pragma unroll
      for (int e = 0; e < 8; ++e) {
        int kidx = kb + j * 16 + r;
        int qi = qw + e + 8 * hi;
        if (kidx > qi + P_LEN) s[j][e] = MASK_NEG;
      }

    // Online softmax per row (stats uniform across each 16-lane group)
#pragma unroll
    for (int e = 0; e < 8; ++e) {
      float t0 = fmaxf(fmaxf(s[0][e], s[1][e]), fmaxf(s[2][e], s[3][e]));
#pragma unroll
      for (int off = 1; off < 16; off <<= 1) t0 = fmaxf(t0, __shfl_xor(t0, off));
      float nm = fmaxf(mrun[e], t0);
      float corr = __expf(mrun[e] - nm);
      mrun[e] = nm;
      float ss = 0.0f;
#pragma unroll
      for (int j = 0; j < 4; ++j) {
        float p = __expf(s[j][e] - nm);
        s[j][e] = p;
        ss += p;
      }
#pragma unroll
      for (int off = 1; off < 16; off <<= 1) ss += __shfl_xor(ss, off);
      lrun[e] = lrun[e] * corr + ss;
#pragma unroll
      for (int t = 0; t < 4; ++t) o[t][e] *= corr;
    }

    // C-layout -> A-layout via wave-private LDS bounce
    _Float16* pw = &Ps[w * 16 * 72];
#pragma unroll
    for (int j = 0; j < 4; ++j)
#pragma unroll
      for (int e = 0; e < 8; ++e)
        pw[(e + 8 * hi) * 72 + j * 16 + r] = (_Float16)s[j][e];

    // O += P (16x64) * V (64x64); B^T = Vt
    v16h pa0 = load_frag(pw, 72);
    v16h pa1 = load_frag(pw + 32, 72);
#pragma unroll
    for (int t = 0; t < 4; ++t) {
      v16h b0 = load_frag(&VtB[t * 16 * 72], 72);
      o[t] = wmma_f16(pa0, b0, o[t]);
      v16h b1 = load_frag(&VtB[t * 16 * 72 + 32], 72);
      o[t] = wmma_f16(pa1, b1, o[t]);
    }
    __syncthreads();
  }

  // Normalize and emit (S, E) f16 for the projection GEMM
#pragma unroll
  for (int t = 0; t < 4; ++t)
#pragma unroll
    for (int e = 0; e < 8; ++e) {
      int qi = qw + e + 8 * hi;
      float val = o[t][e] / lrun[e];
      a_h[(size_t)qi * E_DIM + h * D_DIM + t * 16 + r] = (_Float16)val;
    }
}

// ---------------------------------------------------------------------------
// Host-side launch
// ---------------------------------------------------------------------------
extern "C" void kernel_launch(void* const* d_in, const int* in_sizes, int n_in,
                              void* d_out, int out_size, void* d_ws, size_t ws_size,
                              hipStream_t stream) {
  (void)in_sizes; (void)n_in; (void)out_size; (void)ws_size;
  const float* x       = (const float*)d_in[0];
  const float* past_k  = (const float*)d_in[1];
  const float* past_v  = (const float*)d_in[2];
  /* mask (d_in[3]) is analytic (k <= q + P) -> never read */
  const float* w_attn  = (const float*)d_in[4];
  const float* b_attn  = (const float*)d_in[5];
  const float* w_proj  = (const float*)d_in[6];
  const float* b_proj  = (const float*)d_in[7];

  float* outA  = (float*)d_out;
  float* presK = outA + (size_t)S_LEN * E_DIM;
  float* presV = presK + (size_t)H_NUM * KV_LEN * D_DIM;

  _Float16* xh     = (_Float16*)d_ws;
  _Float16* wattnT = xh + (size_t)S_LEN * E_DIM;
  _Float16* wprojT = wattnT + (size_t)3 * E_DIM * E_DIM;
  _Float16* q_h    = wprojT + (size_t)E_DIM * E_DIM;
  _Float16* k_h    = q_h + (size_t)H_NUM * S_LEN * D_DIM;
  _Float16* v_h    = k_h + (size_t)H_NUM * KV_LEN * D_DIM;
  _Float16* a_h    = v_h + (size_t)H_NUM * KV_LEN * D_DIM;

  prep_kernel<<<4096, 256, 0, stream>>>(x, w_attn, w_proj, past_k, past_v,
                                        xh, wattnT, wprojT, k_h, v_h, presK, presV);

  gemm_kernel<<<dim3(S_LEN / 128, (3 * E_DIM) / 128), 256, 0, stream>>>(
      xh, wattnT, b_attn, E_DIM, /*mode=*/0,
      nullptr, 0, q_h, k_h, v_h, presK, presV);

  attn_kernel<<<dim3(S_LEN / 64, H_NUM), 128, 0, stream>>>(q_h, k_h, v_h, a_h);

  gemm_kernel<<<dim3(S_LEN / 128, E_DIM / 128), 256, 0, stream>>>(
      a_h, wprojT, b_proj, E_DIM, /*mode=*/1,
      outA, E_DIM, nullptr, nullptr, nullptr, nullptr, nullptr);
}
